// FastTransformerModule_44401371906400
// MI455X (gfx1250) — matrix operand order repaired and checked by reading, not compiled, not run on hardware
//
#include <hip/hip_runtime.h>
#include <hip/hip_bf16.h>
#include <cstdint>

#define EMBED 512
#define HEADS 8
#define HD 64
#define LAYERS 4
#define SDIM 3
#define BATCH 2
#define SEQ 1024
#define NS (BATCH * SEQ)      // 2048 rows
#define FFD (4 * EMBED)       // 2048
#define CH 64                 // attention chunk
#define NCH (SEQ / CH)        // 16 chunks
#define BHN (BATCH * HEADS)   // 16
#define VOCABP1 17
#define LN_EPS 1e-5f
#define ATT_EPS 1e-6f

typedef __attribute__((ext_vector_type(16))) __bf16 v16bf;
typedef __attribute__((ext_vector_type(8)))  float  v8f;
typedef __attribute__((ext_vector_type(4)))  unsigned int u32x4;
typedef __attribute__((ext_vector_type(8)))  int i32x8;
typedef __attribute__((ext_vector_type(4)))  int i32x4;

#if defined(__HIP_DEVICE_COMPILE__) && __has_builtin(__builtin_amdgcn_tensor_load_to_lds) && __has_builtin(__builtin_amdgcn_s_wait_tensorcnt)
#define USE_TDM 1
#else
#define USE_TDM 0
#endif

__device__ __forceinline__ __bf16 f2bf(float f) {
  union { float f; unsigned u; } a; a.f = f;
  unsigned u = a.u + 0x7FFFu + ((a.u >> 16) & 1u);   // round-to-nearest-even
  union { unsigned short s; __bf16 b; } r; r.s = (unsigned short)(u >> 16);
  return r.b;
}

// ---------------------------------------------------------------- conversions
__global__ __launch_bounds__(256) void cvt_f32_bf16(const float* __restrict__ s,
                                                    __bf16* __restrict__ d, int n) {
  int i = blockIdx.x * 256 + threadIdx.x;
  if (i < n) d[i] = f2bf(s[i]);
}

// Transpose+convert: W[l][K][N] (f32) -> Wt[l][N][K] (bf16), tiled through LDS.
__global__ __launch_bounds__(256) void cvt_transpose_bf16(
    const float* __restrict__ W, __bf16* __restrict__ Wt, int K, int N) {
  __shared__ float tile[32][33];
  int l = blockIdx.z;
  int k0 = blockIdx.x * 32, n0 = blockIdx.y * 32;
  int tx = threadIdx.x & 31, ty = threadIdx.x >> 5;   // 32 x 8
  const float* Wl = W + (size_t)l * K * N;
  __bf16* Wtl = Wt + (size_t)l * K * N;
  #pragma unroll
  for (int i = 0; i < 32; i += 8)
    tile[ty + i][tx] = Wl[(size_t)(k0 + ty + i) * N + n0 + tx];
  __syncthreads();
  #pragma unroll
  for (int i = 0; i < 32; i += 8)
    Wtl[(size_t)(n0 + ty + i) * K + k0 + tx] = f2bf(tile[tx][ty + i]);
}

// ---------------------------------------------------------------- embedding
__global__ __launch_bounds__(256) void embed_kernel(
    const int* __restrict__ value, const int* __restrict__ depth, const int* __restrict__ pos,
    const float* __restrict__ sos, const float* __restrict__ tok,
    const float* __restrict__ dep, const float* __restrict__ spat,
    float* __restrict__ x, __bf16* __restrict__ xb, float* __restrict__ kmask) {
  int row = blockIdx.x;                  // n*SEQ + s
  int s = row % SEQ;
  int tid = threadIdx.x;
  if (tid == 0) kmask[row] = (value[row] != 0) ? 1.f : 0.f;
  size_t base = (size_t)row * EMBED;
  for (int e = tid; e < EMBED; e += 256) {
    float val;
    if (s == 0) {
      val = sos[e];
    } else {
      int p = row - 1;  // same batch row, position s-1
      val = tok[(size_t)value[p] * EMBED + e] + dep[(size_t)depth[p] * EMBED + e];
      #pragma unroll
      for (int a = 0; a < SDIM; a++)
        val += spat[((size_t)a * 257 + pos[a * NS + p]) * EMBED + e];
    }
    x[base + e] = val;
    xb[base + e] = f2bf(val);
  }
}

#if USE_TDM
// Issue one 2D TDM tile load: rows of 32 bf16 (64B), LDS row stride 96B via
// pad_interval=3 (16 dwords) + pad_amount=7 (8 dwords).  D# per ISA 08 §8.
__device__ __forceinline__ void tdm_issue_tile(unsigned long long gaddr, unsigned lds_off,
                                               int dim0 /*K*/, int dim1 /*tensor rows*/,
                                               int tile1 /*tile rows*/) {
  u32x4 g0;
  g0[0] = 1u;                                                    // count=1 (valid)
  g0[1] = lds_off;                                               // lds_addr (bytes)
  g0[2] = (unsigned)gaddr;                                       // global_addr[31:0]
  g0[3] = (unsigned)((gaddr >> 32) & 0x1FFFFFFull) | (2u << 30); // addr[56:32] | type=2
  i32x8 g1;
  g1[0] = (1 << 16) | (1 << 20) | (3 << 22) | (7 << 25);  // 2B elems, pad 64B+32B
  g1[1] = (dim0 & 0xFFFF) << 16;                          // tensor_dim0 lo16 @bit48
  g1[2] = ((unsigned)dim0 >> 16) | ((unsigned)(dim1 & 0xFFFF) << 16);
  g1[3] = ((unsigned)dim1 >> 16) | (32u << 16);           // tile_dim0 = 32 elems
  g1[4] = tile1;                                          // tile_dim1 = rows
  g1[5] = dim0;                                           // tensor_dim0_stride = K
  g1[6] = 0;
  g1[7] = 0;
  i32x4 z4 = {0, 0, 0, 0};
  i32x8 z8 = {0, 0, 0, 0, 0, 0, 0, 0};
  __builtin_amdgcn_tensor_load_to_lds(g0, g1, z4, z4, z8, 0);
}
#endif

// ---------------------------------------------------------------- WMMA GEMM
// C[M,N] = act(A[M,K] @ Wt[N,K]^T + bias[N]);  A,Wt bf16 (K-contiguous), C f32.
// act: 0=none, 1=elu+1, 2=elu+1 * mask[row], 3=gelu(exact)
// Dynamic LDS: A double buffer 2x(128x48 bf16) then B double buffer 2x(64x48).
#define GEMM_LDS_BYTES (2 * 128 * 48 * 2 + 2 * 64 * 48 * 2)   // 36864
__global__ __launch_bounds__(256) void gemm_bf16_kernel(
    const __bf16* __restrict__ A, const __bf16* __restrict__ Wt,
    const float* __restrict__ bias, const float* __restrict__ mask,
    float* __restrict__ C, __bf16* __restrict__ Cb,
    int M, int K, int N, int act) {
  extern __shared__ __bf16 smem[];
  __bf16* AsBase = smem;                 // buf stride 6144 bf16 (12288 B)
  __bf16* BsBase = smem + 12288;         // buf stride 3072 bf16 (6144 B)
  const int tid  = threadIdx.x;
  const int lane = tid & 31;
  const int wave = tid >> 5;
  const int half = lane >> 4;
  const int r    = lane & 15;
  const int wm   = (wave >> 1) << 5;     // 0,32,64,96
  const int wn   = (wave & 1) << 5;      // 0,32
  const int m0   = blockIdx.y * 128;
  const int n0   = blockIdx.x * 64;
  const int T    = K >> 5;

  v8f acc[2][2] = {};

#if USE_TDM
  const unsigned long long gA = (unsigned long long)(A  + (size_t)m0 * K);
  const unsigned long long gB = (unsigned long long)(Wt + (size_t)n0 * K);
  if (wave == 0) {
    tdm_issue_tile(gA, 0u, K, M, 128);
    tdm_issue_tile(gB, 24576u, K, N, 64);
  }
#else
  const int arow = tid >> 1, acol = (tid & 1) << 4;   // A: 128 rows x 32, 16 bf16/thread
  const int brow = tid >> 2, bcol = (tid & 3) << 3;   // B: 64 rows x 32, 8 bf16/thread
  const __bf16* apg = A  + (size_t)(m0 + arow) * K + acol;
  const __bf16* bpg = Wt + (size_t)(n0 + brow) * K + bcol;
  u32x4 rA0 = *(const u32x4*)apg;
  u32x4 rA1 = *(const u32x4*)(apg + 8);
  u32x4 rB0 = *(const u32x4*)bpg;
#endif

  for (int t = 0; t < T; t++) {
    __bf16* As = AsBase + (t & 1) * 6144;
    __bf16* Bs = BsBase + (t & 1) * 3072;
#if USE_TDM
    if (wave == 0) __builtin_amdgcn_s_wait_tensorcnt(0);
#else
    *(u32x4*)(As + arow * 48 + acol)     = rA0;
    *(u32x4*)(As + arow * 48 + acol + 8) = rA1;
    *(u32x4*)(Bs + brow * 48 + bcol)     = rB0;
#endif
    __syncthreads();
    if (t + 1 < T) {
#if USE_TDM
      if (wave == 0) {
        unsigned nb = (unsigned)((t + 1) & 1);
        unsigned long long koff = (unsigned long long)(t + 1) * 64ull;  // 32 elems * 2B
        tdm_issue_tile(gA + koff, nb * 12288u,          K, M, 128);
        tdm_issue_tile(gB + koff, 24576u + nb * 6144u,  K, N, 64);
      }
#else
      const __bf16* ap = apg + (t + 1) * 32;
      rA0 = *(const u32x4*)ap;
      rA1 = *(const u32x4*)(ap + 8);
      rB0 = *(const u32x4*)(bpg + (t + 1) * 32);
#endif
    }
    union Frag { v16bf v; u32x4 q[2]; } af[2], bfm[2];
    #pragma unroll
    for (int mi = 0; mi < 2; mi++) {
      const __bf16* rp = As + (wm + (mi << 4) + r) * 48;
      af[mi].q[0] = *(const u32x4*)(rp + (half << 3));        // K 0..7  / 8..15
      af[mi].q[1] = *(const u32x4*)(rp + 16 + (half << 3));   // K 16..23 / 24..31
    }
    #pragma unroll
    for (int ni = 0; ni < 2; ni++) {
      const __bf16* rp = Bs + (wn + (ni << 4) + r) * 48;
      bfm[ni].q[0] = *(const u32x4*)(rp + (half << 3));       // K 0..7  / 8..15
      bfm[ni].q[1] = *(const u32x4*)(rp + 16 + (half << 3));  // K 16..23 / 24..31
    }
    #pragma unroll
    for (int mi = 0; mi < 2; mi++)
      #pragma unroll
      for (int ni = 0; ni < 2; ni++)
        acc[mi][ni] = __builtin_amdgcn_wmma_f32_16x16x32_bf16(
            false, af[mi].v, false, bfm[ni].v, (short)0, acc[mi][ni], false, false);
  }

  // epilogue: C/D layout -> VGPR i holds M = i + 8*half, N = r
  #pragma unroll
  for (int mi = 0; mi < 2; mi++)
    #pragma unroll
    for (int ni = 0; ni < 2; ni++) {
      int gcol = n0 + wn + (ni << 4) + r;
      float bv = bias[gcol];
      #pragma unroll
      for (int i = 0; i < 8; i++) {
        int grow = m0 + wm + (mi << 4) + (half << 3) + i;
        float xv = acc[mi][ni][i] + bv;
        if (act == 1)      xv = xv > 0.f ? xv + 1.f : __expf(xv);
        else if (act == 2) { xv = xv > 0.f ? xv + 1.f : __expf(xv); xv *= mask[grow]; }
        else if (act == 3) xv = 0.5f * xv * (1.f + erff(xv * 0.70710678118654752f));
        size_t o = (size_t)grow * N + gcol;
        C[o] = xv;
        if (Cb) Cb[o] = f2bf(xv);
      }
    }
}

// ------------------------------------------------- attention: per-chunk sums
__global__ __launch_bounds__(256) void attn_sums_kernel(
    const float* __restrict__ k, const float* __restrict__ v,
    float* __restrict__ Kv, float* __restrict__ Ks) {
  __shared__ float sk[CH][HD + 1], sv[CH][HD + 1];
  int c = blockIdx.x, bh = blockIdx.y;
  int b = bh >> 3, h = bh & 7;
  int tid = threadIdx.x;
  size_t rowbase = (size_t)(b * SEQ + c * CH);
  for (int idx = tid; idx < CH * HD; idx += 256) {
    int j = idx >> 6, d = idx & 63;
    size_t g = (rowbase + j) * EMBED + h * HD + d;
    sk[j][d] = k[g]; sv[j][d] = v[g];
  }
  __syncthreads();
  int d = tid >> 2, m0 = (tid & 3) << 4;
  float accv[16];
  #pragma unroll
  for (int m = 0; m < 16; m++) accv[m] = 0.f;
  for (int j = 0; j < CH; j++) {
    float kd = sk[j][d];
    #pragma unroll
    for (int m = 0; m < 16; m++) accv[m] += kd * sv[j][m0 + m];
  }
  float* outp = Kv + ((size_t)(bh * NCH + c)) * (HD * HD) + d * HD + m0;
  #pragma unroll
  for (int m = 0; m < 16; m++) outp[m] = accv[m];
  if (tid < HD) {
    float s = 0.f;
    for (int j = 0; j < CH; j++) s += sk[j][tid];
    Ks[(bh * NCH + c) * HD + tid] = s;
  }
}

// ------------------------------------------------- attention: exclusive scan
__global__ __launch_bounds__(256) void attn_scan_kernel(float* __restrict__ Kv,
                                                        float* __restrict__ Ks) {
  int bh = blockIdx.x, tid = threadIdx.x;
  for (int e = tid; e < HD * HD; e += 256) {
    float run = 0.f;
    for (int c = 0; c < NCH; c++) {
      float* p = Kv + ((size_t)(bh * NCH + c)) * (HD * HD) + e;
      float t = *p; *p = run; run += t;
    }
  }
  if (tid < HD) {
    float run = 0.f;
    for (int c = 0; c < NCH; c++) {
      float* p = Ks + (bh * NCH + c) * HD + tid;
      float t = *p; *p = run; run += t;
    }
  }
}

// ------------------------------------------------- attention: chunk outputs
__global__ __launch_bounds__(256) void attn_out_kernel(
    const float* __restrict__ q, const float* __restrict__ k, const float* __restrict__ v,
    const float* __restrict__ Kv, const float* __restrict__ Ks,
    float* __restrict__ numz, __bf16* __restrict__ numzb) {
  __shared__ float sq[CH][HD + 1], sk[CH][HD + 1], sv[CH][HD + 1];
  __shared__ float sS[HD][HD + 1], sA[CH][CH + 1];
  __shared__ float skp[HD], sden[CH];
  int c = blockIdx.x, bh = blockIdx.y;
  int b = bh >> 3, h = bh & 7;
  int tid = threadIdx.x;
  size_t rowbase = (size_t)(b * SEQ + c * CH);
  const float* Sg = Kv + ((size_t)(bh * NCH + c)) * (HD * HD);
  for (int idx = tid; idx < CH * HD; idx += 256) {
    int j = idx >> 6, d = idx & 63;
    size_t g = (rowbase + j) * EMBED + h * HD + d;
    sq[j][d] = q[g]; sk[j][d] = k[g]; sv[j][d] = v[g];
    sS[j][d] = Sg[idx];
  }
  if (tid < HD) skp[tid] = Ks[(bh * NCH + c) * HD + tid];
  __syncthreads();
  // A = tril(q k^T)
  int i = tid >> 2, j0 = (tid & 3) << 4;
  for (int jj = 0; jj < 16; jj++) {
    int j = j0 + jj;
    float s = 0.f;
    if (j <= i) { for (int d = 0; d < HD; d++) s += sq[i][d] * sk[j][d]; }
    sA[i][j] = s;
  }
  __syncthreads();
  // denominator: rowsum(A_i) + q_i . Ks_prev
  if (tid < CH) {
    float s = 0.f;
    for (int j = 0; j <= tid; j++) s += sA[tid][j];
    for (int d = 0; d < HD; d++) s += sq[tid][d] * skp[d];
    sden[tid] = 1.f / (s + ATT_EPS);
  }
  __syncthreads();
  // numerator: A @ v + q @ S_prev
  int m0 = j0;
  float accv[16];
  #pragma unroll
  for (int m = 0; m < 16; m++) accv[m] = 0.f;
  for (int j = 0; j < CH; j++) {
    float aij = sA[i][j];
    #pragma unroll
    for (int m = 0; m < 16; m++) accv[m] += aij * sv[j][m0 + m];
  }
  for (int d = 0; d < HD; d++) {
    float qd = sq[i][d];
    #pragma unroll
    for (int m = 0; m < 16; m++) accv[m] += qd * sS[d][m0 + m];
  }
  float z = sden[i];
  size_t ob = (rowbase + i) * EMBED + h * HD + m0;
  #pragma unroll
  for (int m = 0; m < 16; m++) {
    float val = accv[m] * z;
    numz[ob + m] = val;
    numzb[ob + m] = f2bf(val);
  }
}

// ---------------------------------------------------------------- residual+LN
__global__ __launch_bounds__(256) void ln_kernel(
    float* __restrict__ x, const float* __restrict__ add,
    const float* __restrict__ g, const float* __restrict__ b, __bf16* __restrict__ xb) {
  __shared__ float red[256];
  int row = blockIdx.x, tid = threadIdx.x;
  size_t base = (size_t)row * EMBED;
  float v0 = x[base + tid];
  float v1 = x[base + 256 + tid];
  if (add) { v0 += add[base + tid]; v1 += add[base + 256 + tid]; }
  red[tid] = v0 + v1;
  __syncthreads();
  for (int s = 128; s > 0; s >>= 1) { if (tid < s) red[tid] += red[tid + s]; __syncthreads(); }
  float mean = red[0] * (1.f / EMBED);
  __syncthreads();
  float d0 = v0 - mean, d1 = v1 - mean;
  red[tid] = d0 * d0 + d1 * d1;
  __syncthreads();
  for (int s = 128; s > 0; s >>= 1) { if (tid < s) red[tid] += red[tid + s]; __syncthreads(); }
  float rstd = rsqrtf(red[0] * (1.f / EMBED) + LN_EPS);
  float o0 = d0 * rstd * g[tid] + b[tid];
  float o1 = d1 * rstd * g[256 + tid] + b[256 + tid];
  x[base + tid] = o0;
  x[base + 256 + tid] = o1;
  if (xb) { xb[base + tid] = f2bf(o0); xb[base + 256 + tid] = f2bf(o1); }
}

// ---------------------------------------------------------------- head GEMV
__global__ __launch_bounds__(256) void head_kernel(
    const float* __restrict__ x, const float* __restrict__ Wh, float* __restrict__ out) {
  int row  = blockIdx.x * 8 + (threadIdx.x >> 5);
  int vtok = threadIdx.x & 31;
  if (vtok >= VOCABP1) return;
  float s = 0.f;
  size_t base = (size_t)row * EMBED;
  for (int e = 0; e < EMBED; e++) s += x[base + e] * Wh[(size_t)e * VOCABP1 + vtok];
  out[(size_t)row * VOCABP1 + vtok] = s;
}

// ---------------------------------------------------------------- launch
extern "C" void kernel_launch(void* const* d_in, const int* in_sizes, int n_in,
                              void* d_out, int out_size, void* d_ws, size_t ws_size,
                              hipStream_t stream) {
  (void)in_sizes; (void)n_in; (void)out_size; (void)ws_size;
  const int*   value = (const int*)d_in[0];
  const int*   depth = (const int*)d_in[1];
  const int*   pos   = (const int*)d_in[2];
  const float* sos   = (const float*)d_in[3];
  const float* tok   = (const float*)d_in[4];
  const float* dep   = (const float*)d_in[5];
  const float* spat  = (const float*)d_in[6];
  const float* Wq = (const float*)d_in[7];
  const float* Wk = (const float*)d_in[8];
  const float* Wv = (const float*)d_in[9];
  const float* Wo = (const float*)d_in[10];
  const float* bq = (const float*)d_in[11];
  const float* bk = (const float*)d_in[12];
  const float* bv = (const float*)d_in[13];
  const float* bo = (const float*)d_in[14];
  const float* ln1g = (const float*)d_in[15];
  const float* ln1b = (const float*)d_in[16];
  const float* ln2g = (const float*)d_in[17];
  const float* ln2b = (const float*)d_in[18];
  const float* W1 = (const float*)d_in[19];
  const float* b1 = (const float*)d_in[20];
  const float* W2 = (const float*)d_in[21];
  const float* b2 = (const float*)d_in[22];
  const float* lnfg = (const float*)d_in[23];
  const float* lnfb = (const float*)d_in[24];
  const float* Wh   = (const float*)d_in[25];
  float* out = (float*)d_out;

  char* p = (char*)d_ws;
  auto alloc = [&](size_t bytes) -> void* {
    void* r = (void*)p;
    p += (bytes + 255) & ~(size_t)255;
    return r;
  };

  const size_t EE = (size_t)EMBED * EMBED;
  const size_t EF = (size_t)EMBED * FFD;
  __bf16* WqT = (__bf16*)alloc(LAYERS * EE * 2);   // [l][N=E][K=E]
  __bf16* WkT = (__bf16*)alloc(LAYERS * EE * 2);
  __bf16* WvT = (__bf16*)alloc(LAYERS * EE * 2);
  __bf16* WoT = (__bf16*)alloc(LAYERS * EE * 2);
  __bf16* W1T = (__bf16*)alloc(LAYERS * EF * 2);   // [l][N=FF][K=E]
  __bf16* W2T = (__bf16*)alloc(LAYERS * EF * 2);   // [l][N=E][K=FF]
  float*  x     = (float*) alloc((size_t)NS * EMBED * 4);
  __bf16* xb    = (__bf16*)alloc((size_t)NS * EMBED * 2);
  float*  kmask = (float*) alloc((size_t)NS * 4);
  float*  qbuf  = (float*) alloc((size_t)NS * EMBED * 4);
  float*  kbuf  = (float*) alloc((size_t)NS * EMBED * 4);
  float*  vbuf  = (float*) alloc((size_t)NS * EMBED * 4);
  float*  Kv    = (float*) alloc((size_t)BHN * NCH * HD * HD * 4);
  float*  Ks    = (float*) alloc((size_t)BHN * NCH * HD * 4);
  float*  numz  = (float*) alloc((size_t)NS * EMBED * 4);
  __bf16* numzb = (__bf16*)alloc((size_t)NS * EMBED * 2);
  float*  attnb = (float*) alloc((size_t)NS * EMBED * 4);
  float*  hbuf  = (float*) alloc((size_t)NS * FFD * 4);
  __bf16* hb    = (__bf16*)alloc((size_t)NS * FFD * 2);
  float*  ybuf  = (float*) alloc((size_t)NS * EMBED * 4);

  // weight transpose+convert: [K][N] f32 -> [N][K] bf16
  const dim3 tEE(EMBED / 32, EMBED / 32, LAYERS);
  cvt_transpose_bf16<<<tEE, 256, 0, stream>>>(Wq, WqT, EMBED, EMBED);
  cvt_transpose_bf16<<<tEE, 256, 0, stream>>>(Wk, WkT, EMBED, EMBED);
  cvt_transpose_bf16<<<tEE, 256, 0, stream>>>(Wv, WvT, EMBED, EMBED);
  cvt_transpose_bf16<<<tEE, 256, 0, stream>>>(Wo, WoT, EMBED, EMBED);
  cvt_transpose_bf16<<<dim3(EMBED / 32, FFD / 32, LAYERS), 256, 0, stream>>>(W1, W1T, EMBED, FFD);
  cvt_transpose_bf16<<<dim3(FFD / 32, EMBED / 32, LAYERS), 256, 0, stream>>>(W2, W2T, FFD, EMBED);

  embed_kernel<<<NS, 256, 0, stream>>>(value, depth, pos, sos, tok, dep, spat, x, xb, kmask);

  const dim3 gE(EMBED / 64, NS / 128);   // (8,16)  N=512 GEMMs
  const dim3 gF(FFD / 64,   NS / 128);   // (32,16) N=2048 GEMM
  const dim3 gA(NCH, BHN);               // (16,16) attention chunk grids
  const unsigned SH = GEMM_LDS_BYTES;

  for (int i = 0; i < LAYERS; i++) {
    const __bf16* wq = WqT + (size_t)i * EE;
    const __bf16* wk = WkT + (size_t)i * EE;
    const __bf16* wv = WvT + (size_t)i * EE;
    const __bf16* wo = WoT + (size_t)i * EE;
    const __bf16* w1 = W1T + (size_t)i * EF;
    const __bf16* w2 = W2T + (size_t)i * EF;

    gemm_bf16_kernel<<<gE, 256, SH, stream>>>(xb, wq, bq + i * EMBED, nullptr,
                                              qbuf, nullptr, NS, EMBED, EMBED, 1);
    gemm_bf16_kernel<<<gE, 256, SH, stream>>>(xb, wk, bk + i * EMBED, kmask,
                                              kbuf, nullptr, NS, EMBED, EMBED, 2);
    gemm_bf16_kernel<<<gE, 256, SH, stream>>>(xb, wv, bv + i * EMBED, nullptr,
                                              vbuf, nullptr, NS, EMBED, EMBED, 0);

    attn_sums_kernel<<<gA, 256, 0, stream>>>(kbuf, vbuf, Kv, Ks);
    attn_scan_kernel<<<BHN, 256, 0, stream>>>(Kv, Ks);
    attn_out_kernel<<<gA, 256, 0, stream>>>(qbuf, kbuf, vbuf, Kv, Ks, numz, numzb);

    gemm_bf16_kernel<<<gE, 256, SH, stream>>>(numzb, wo, bo + i * EMBED, nullptr,
                                              attnb, nullptr, NS, EMBED, EMBED, 0);
    ln_kernel<<<NS, 256, 0, stream>>>(x, attnb, ln1g + i * EMBED, ln1b + i * EMBED, xb);

    gemm_bf16_kernel<<<gF, 256, SH, stream>>>(xb, w1, b1 + i * FFD, nullptr,
                                              hbuf, hb, NS, EMBED, FFD, 3);
    gemm_bf16_kernel<<<gE, 256, SH, stream>>>(hb, w2, b2 + i * EMBED, nullptr,
                                              ybuf, nullptr, NS, FFD, EMBED, 0);
    ln_kernel<<<NS, 256, 0, stream>>>(x, ybuf, ln2g + i * EMBED, ln2b + i * EMBED, xb);
  }

  ln_kernel<<<NS, 256, 0, stream>>>(x, nullptr, lnfg, lnfb, nullptr);
  head_kernel<<<NS / 8, 256, 0, stream>>>(x, Wh, out);
}
